// MSBGHearingModel_65369402245820
// MI455X (gfx1250) — matrix operand order repaired and checked by reading, not compiled, not run on hardware
//
#include <hip/hip_runtime.h>
#include <hip/hip_fp16.h>
#include <math.h>

// ---------------------------------------------------------------------------
// MSBG hearing model on gfx1250 (MI455X).
// Heavy stages (gammatone filterbank conv, STFT/smear/iSTFT GEMMs) run on the
// f16 WMMA pipe (v_wmma_f32_16x16x32_f16, f32 accumulate); scalar stages
// (calibration, 1015-tap FIRs, grouped LPF, OLA, aligned sum) are VALU.
// ---------------------------------------------------------------------------

typedef _Float16 h1;
typedef __attribute__((ext_vector_type(8)))  _Float16 h8;
typedef __attribute__((ext_vector_type(16))) _Float16 v16h;
typedef __attribute__((ext_vector_type(8)))  float    v8f;

#define EPSV 1e-8f
#define B_N   4
#define T_LEN 131072            // 2^17
#define NC    1015              // cochlea FIR taps
#define PADC  507
#define KGT   1025              // gammatone taps
#define TR    (T_LEN + KGT - 1) // 132096
#define C_CH  30
#define C32   32
#define WINL  256
#define HOPL  64
#define NFR   2045              // (T-256)/64+1
#define NBN   (B_N * NFR)       // 8180
#define MT    512               // ceil(NBN/16)
#define NT_STFT 33              // ceil(514/16)
#define KINV  544               // 257 re + 257 im + pad
#define LPAD  (1024 + T_LEN + 1056)
#define RMSW  441
#define NFRM  297               // T // 441
#define TWOPI 6.283185307179586f

__device__ __forceinline__ v8f wmma_f16(v16h a, v16h b, v8f c) {
  return __builtin_amdgcn_wmma_f32_16x16x32_f16(false, a, false, b, (short)0, c,
                                                false, false);
}

__device__ __forceinline__ v16h make_a(h8 lo, h8 hi) {
  v16h a;
#pragma unroll
  for (int i = 0; i < 8; ++i) { a[i] = lo[i]; a[i + 8] = hi[i]; }
  return a;
}

__device__ __forceinline__ float block_reduce_sum(float v, float* sm) {
  int tid = threadIdx.x;
  sm[tid] = v; __syncthreads();
#pragma unroll
  for (int s = 128; s > 0; s >>= 1) {
    if (tid < s) sm[tid] += sm[tid + s];
    __syncthreads();
  }
  float r = sm[0]; __syncthreads();
  return r;
}

// --------------------------- 1. SPL calibration -----------------------------
__global__ void k_calibrate(const float* __restrict__ x, float* __restrict__ gain) {
  __shared__ float sm[256];
  int b = blockIdx.x, tid = threadIdx.x;
  const float* xb = x + (size_t)b * T_LEN;
  float s2 = 0.f;
  for (int t = tid; t < T_LEN; t += 256) { float v = xb[t]; s2 += v * v; }
  float sumsq = block_reduce_sum(s2, sm);
  float msq = sumsq / (float)T_LEN;
  float thr = 20.f * log10f(sqrtf(msq + EPSV) + EPSV) - 12.f;
  float pk = 0.f, kc = 0.f;
  for (int f = tid; f < NFRM; f += 256) {
    const float* fr = xb + f * RMSW;
    float s = 0.f;
    for (int i = 0; i < RMSW; ++i) { float v = fr[i]; s += v * v; }
    float pw = s / (float)RMSW;
    float key = (10.f * log10f(pw + EPSV) > thr) ? 1.f : 0.f;
    pk += pw * key; kc += key;
  }
  float spk = block_reduce_sum(pk, sm);
  float skc = block_reduce_sum(kc, sm);
  if (tid == 0) {
    float rms = sqrtf(spk / (skc + EPSV) + EPSV);
    float change = (120.f + 10.f * log10f(msq + EPSV)) -
                   (120.f + 20.f * log10f(rms + EPSV));
    gain[b] = __powf(10.f, 0.05f * change);
  }
}

// ---------------- 2. cochlea fwd FIR (gain folded in), f32 -> f16 -----------
__global__ void k_conv_fwd(const float* __restrict__ x, const float* __restrict__ w,
                           const float* __restrict__ gain, h1* __restrict__ out) {
  __shared__ float wt[NC];
  __shared__ float xs[256 + NC - 1];
  int b = blockIdx.x >> 9;
  int t0 = (blockIdx.x & 511) << 8;
  int tid = threadIdx.x;
  const float* xb = x + (size_t)b * T_LEN;
  float g = gain[b];
  for (int i = tid; i < NC; i += 256) wt[i] = w[i];
  for (int i = tid; i < 256 + NC - 1; i += 256) {
    int t = t0 - PADC + i;
    xs[i] = (t >= 0 && t < T_LEN) ? xb[t] * g : 0.f;
  }
  __syncthreads();
  float acc = 0.f;
  for (int k = 0; k < NC; ++k) acc = fmaf(wt[k], xs[tid + k], acc);
  out[(size_t)b * T_LEN + t0 + tid] = (h1)acc;
}

// ------------------------ 3. fragment-pack init kernels ---------------------
// B-fragment layout (32x16 f16): lane l -> n = l%16, K = 8*(l/16) + (h&7) + 16*(h>>3), h=0..15
__global__ void k_pack_gt(const float* __restrict__ gt, const float* __restrict__ gts,
                          h1* __restrict__ wf) {
  int idx = blockIdx.x * 256 + threadIdx.x;          // 2*2*33*512
  if (idx >= 2 * 2 * 33 * 512) return;
  int h = idx & 15, l = (idx >> 4) & 31, kc = (idx >> 9) % 33, sc = idx / (33 * 512);
  int s = sc >> 1, ct = sc & 1;
  int c = ct * 16 + (l & 15);
  int k = kc * 32 + 8 * (l >> 4) + (h & 7) + 16 * (h >> 3);
  float v = 0.f;
  if (c < C_CH && k < KGT) v = (s == 0 ? gt : gts)[c * KGT + k];
  wf[idx] = (h1)v;
}

__global__ void k_pack_dft(const float* __restrict__ win, h1* __restrict__ bd) {
  int idx = blockIdx.x * 256 + threadIdx.x;          // 8*33*512
  if (idx >= 8 * NT_STFT * 512) return;
  int h = idx & 15, l = (idx >> 4) & 31;
  int nt = (idx >> 9) % NT_STFT, kc = idx / (NT_STFT * 512);
  int k = kc * 32 + 8 * (l >> 4) + (h & 7) + 16 * (h >> 3); // time 0..255
  int j = nt * 16 + (l & 15);                        // 0..527
  float v = 0.f;
  if (j < 257) {
    int ph = (j * k) & 511;
    v = win[k] * cosf((TWOPI / 512.f) * (float)ph);
  } else if (j < 514) {
    int ph = ((j - 257) * k) & 511;
    v = -win[k] * sinf((TWOPI / 512.f) * (float)ph);
  }
  bd[idx] = (h1)v;
}

__global__ void k_pack_smear(const float* __restrict__ fs, h1* __restrict__ bs) {
  int idx = blockIdx.x * 256 + threadIdx.x;          // 8*16*512
  if (idx >= 8 * 16 * 512) return;
  int h = idx & 15, l = (idx >> 4) & 31, nt = (idx >> 9) & 15, kc = idx >> 13;
  int k = kc * 32 + 8 * (l >> 4) + (h & 7) + 16 * (h >> 3); // f_in
  int n = nt * 16 + (l & 15);                        // f_out
  bs[idx] = (h1)fs[n * 256 + k];                     // einsum 'fk'
}

__global__ void k_pack_inv(const float* __restrict__ win, h1* __restrict__ bi) {
  int idx = blockIdx.x * 256 + threadIdx.x;          // 17*16*512
  if (idx >= 17 * 16 * 512) return;
  int h = idx & 15, l = (idx >> 4) & 31, nt = (idx >> 9) & 15, kc = idx >> 13;
  int kk = kc * 32 + 8 * (l >> 4) + (h & 7) + 16 * (h >> 3); // 0..543
  int t = nt * 16 + (l & 15);
  float v = 0.f;
  if (kk < 257) {
    int f = kk; float s = (f == 0 || f == 256) ? 1.f : 2.f;
    int ph = (f * t) & 511;
    v = s * cosf((TWOPI / 512.f) * (float)ph) * win[t] * (1.f / 512.f);
  } else if (kk < 514) {
    int f = kk - 257; float s = (f == 0 || f == 256) ? 1.f : 2.f;
    int ph = (f * t) & 511;
    v = -s * sinf((TWOPI / 512.f) * (float)ph) * win[t] * (1.f / 512.f);
  }
  bi[idx] = (h1)v;
}

// --------------------- 4. forward STFT as WMMA GEMM -------------------------
__global__ void k_stft(const h1* __restrict__ xh, const h1* __restrict__ bd,
                       float* __restrict__ sre, float* __restrict__ sim) {
  int wave = threadIdx.x >> 5, lane = threadIdx.x & 31;
  int gw = blockIdx.x * 8 + wave;                    // 512*33 waves exactly
  int mt = gw / NT_STFT, nt = gw % NT_STFT;
  int m = lane & 15, hi = lane >> 4;
  int bn = mt * 16 + m;
  int bnc = (bn < NBN) ? bn : (NBN - 1);
  int b = bnc / NFR, n = bnc % NFR;
  const h1* arow = xh + (size_t)b * T_LEN + n * HOPL;
  v8f acc = {};
#pragma unroll
  for (int kc = 0; kc < 8; ++kc) {
    h8 lo = *(const h8*)(arow + kc * 32 + 8 * hi);
    h8 hp = *(const h8*)(arow + kc * 32 + 16 + 8 * hi);
    v16h a = make_a(lo, hp);
    v16h bf = *(const v16h*)(bd + (size_t)(kc * NT_STFT + nt) * 512 + lane * 16);
    acc = wmma_f16(a, bf, acc);
  }
  int j = nt * 16 + (lane & 15);
#pragma unroll
  for (int r = 0; r < 8; ++r) {
    int row = mt * 16 + r + 8 * hi;
    if (row < NBN) {
      if (j < 257)      sre[(size_t)row * 257 + j] = acc[r];
      else if (j < 514) sim[(size_t)row * 257 + (j - 257)] = acc[r];
    }
  }
}

// ------------------------- 5. |spec|^2 (f16 out) ----------------------------
__global__ void k_mag2(const float* __restrict__ sre, const float* __restrict__ sim,
                       h1* __restrict__ m2) {
  int idx = blockIdx.x * 256 + threadIdx.x;          // NBN*256
  int bn = idx >> 8, f = idx & 255;
  float re = sre[(size_t)bn * 257 + f], im = sim[(size_t)bn * 257 + f];
  m2[idx] = (h1)(re * re + im * im);
}

// ---------------------- 6. smearing matrix as WMMA GEMM ---------------------
__global__ void k_smear_mm(const h1* __restrict__ m2, const h1* __restrict__ bs,
                           float* __restrict__ smr) {
  int wave = threadIdx.x >> 5, lane = threadIdx.x & 31;
  int gw = blockIdx.x * 8 + wave;                    // 512*16 waves exactly
  int mt = gw >> 4, nt = gw & 15;
  int m = lane & 15, hi = lane >> 4;
  int bn = mt * 16 + m;
  int bnc = (bn < NBN) ? bn : (NBN - 1);
  const h1* arow = m2 + (size_t)bnc * 256;
  v8f acc = {};
#pragma unroll
  for (int kc = 0; kc < 8; ++kc) {
    h8 lo = *(const h8*)(arow + kc * 32 + 8 * hi);
    h8 hp = *(const h8*)(arow + kc * 32 + 16 + 8 * hi);
    v16h a = make_a(lo, hp);
    v16h bf = *(const v16h*)(bs + (size_t)(kc * 16 + nt) * 512 + lane * 16);
    acc = wmma_f16(a, bf, acc);
  }
  int j = nt * 16 + (lane & 15);
#pragma unroll
  for (int r = 0; r < 8; ++r) {
    int row = mt * 16 + r + 8 * hi;
    if (row < NBN) smr[(size_t)row * 256 + j] = acc[r];
  }
}

// ------------------ 7. ratio & modified spectrum (f16) ----------------------
__global__ void k_ratio(const float* __restrict__ sre, const float* __restrict__ sim,
                        const float* __restrict__ smr, h1* __restrict__ smod) {
  int idx = blockIdx.x * 256 + threadIdx.x;
  if (idx >= NBN * 257) return;
  int bn = idx / 257, f = idx % 257;
  float re = sre[idx], im = sim[idx];
  float ratio = 1.f;
  if (f < 256) {
    float mag = sqrtf(re * re + im * im);
    float s = smr[(size_t)bn * 256 + f];
    ratio = sqrtf(fmaxf(s, 0.f)) / (mag + EPSV);
  }
  smod[(size_t)bn * KINV + f] = (h1)(re * ratio);
  smod[(size_t)bn * KINV + 257 + f] = (h1)(im * ratio);
}

// --------------------- 8. inverse STFT as WMMA GEMM -------------------------
__global__ void k_inv(const h1* __restrict__ smod, const h1* __restrict__ bi,
                      float* __restrict__ rec) {
  int wave = threadIdx.x >> 5, lane = threadIdx.x & 31;
  int gw = blockIdx.x * 8 + wave;                    // 512*16 waves
  int mt = gw >> 4, nt = gw & 15;
  int m = lane & 15, hi = lane >> 4;
  int bn = mt * 16 + m;
  int bnc = (bn < NBN) ? bn : (NBN - 1);
  const h1* arow = smod + (size_t)bnc * KINV;
  v8f acc = {};
#pragma unroll
  for (int kc = 0; kc < 17; ++kc) {
    h8 lo = *(const h8*)(arow + kc * 32 + 8 * hi);
    h8 hp = *(const h8*)(arow + kc * 32 + 16 + 8 * hi);
    v16h a = make_a(lo, hp);
    v16h bf = *(const v16h*)(bi + (size_t)(kc * 16 + nt) * 512 + lane * 16);
    acc = wmma_f16(a, bf, acc);
  }
  int j = nt * 16 + (lane & 15);
#pragma unroll
  for (int r = 0; r < 8; ++r) {
    int row = mt * 16 + r + 8 * hi;
    if (row < NBN) rec[(size_t)row * 256 + j] = acc[r];
  }
}

// --------------------------- 9. overlap-add ---------------------------------
__global__ void k_ola(const float* __restrict__ rec, const float* __restrict__ win,
                      h1* __restrict__ xsm) {
  int idx = blockIdx.x * 256 + threadIdx.x;          // B*T
  int b = idx >> 17, t = idx & (T_LEN - 1);
  int nmax = t / HOPL; if (nmax > NFR - 1) nmax = NFR - 1;
  int nmin = (t >= WINL) ? ((t - WINL) / HOPL + 1) : 0;
  float ola = 0.f, wsum = 0.f;
  for (int n = nmin; n <= nmax; ++n) {
    int tt = t - n * HOPL;                           // 0..255
    float w = win[tt];
    ola += rec[((size_t)b * NFR + n) * 256 + tt];
    wsum += w * w;
  }
  float v = (wsum > 1e-6f) ? ola / wsum : 0.f;
  xsm[(size_t)b * LPAD + 1024 + t] = (h1)v;
}

// ----- 10. gammatone filterbank (real+imag) as implicit-GEMM WMMA conv ------
// A: 16 time rows x 32 taps of padded signal.  B: prepacked filter fragments.
// D tile = [time x channel], stored channel-minor; env = sqrt(re^2+im^2+eps).
__global__ void k_recruit(const h1* __restrict__ xsm, const h1* __restrict__ wf,
                          float* __restrict__ yreal, h1* __restrict__ envh) {
  int wave = threadIdx.x >> 5, lane = threadIdx.x & 31;
  int b = blockIdx.x / (TR / 128);
  int tb = blockIdx.x % (TR / 128);
  int t0 = tb * 128 + wave * 16;
  int m = lane & 15, hi = lane >> 4;
  const h1* xb = xsm + (size_t)b * LPAD + (t0 + m);  // net offset: pad folded in
  v8f a00 = {}, a01 = {}, a10 = {}, a11 = {};
  for (int kc = 0; kc < 33; ++kc) {
    const h1* ap = xb + kc * 32 + 8 * hi;
    h8 lo, hp;
    __builtin_memcpy(&lo, ap, sizeof(h8));           // 2B-aligned sliding window
    __builtin_memcpy(&hp, ap + 16, sizeof(h8));
    v16h a = make_a(lo, hp);
    if (kc + 1 < 33) __builtin_prefetch(ap + 32, 0, 1);
    v16h b00 = *(const v16h*)(wf + (size_t)((0 * 33 + kc) * 512) + lane * 16);
    v16h b01 = *(const v16h*)(wf + (size_t)((1 * 33 + kc) * 512) + lane * 16);
    v16h b10 = *(const v16h*)(wf + (size_t)((2 * 33 + kc) * 512) + lane * 16);
    v16h b11 = *(const v16h*)(wf + (size_t)((3 * 33 + kc) * 512) + lane * 16);
    a00 = wmma_f16(a, b00, a00);   // cos, channels 0..15
    a01 = wmma_f16(a, b01, a01);   // cos, channels 16..31
    a10 = wmma_f16(a, b10, a10);   // sin, channels 0..15
    a11 = wmma_f16(a, b11, a11);   // sin, channels 16..31
  }
  int c = lane & 15;
#pragma unroll
  for (int r = 0; r < 8; ++r) {
    int t = t0 + r + 8 * hi;
    size_t base = ((size_t)b * TR + t) * C32;
    float re0 = a00[r], im0 = a10[r];
    float re1 = a01[r], im1 = a11[r];
    yreal[base + c] = re0;
    yreal[base + 16 + c] = re1;
    envh[base + c] = (h1)sqrtf(re0 * re0 + im0 * im0 + EPSV);
    envh[base + 16 + c] = (h1)sqrtf(re1 * re1 + im1 * im1 + EPSV);
  }
}

// -------- 11. grouped envelope LPF + compressive gain (in-place on y) -------
__global__ void k_lpf_gain(const h1* __restrict__ envh, const float* __restrict__ lpf,
                           const float* __restrict__ em1, const float* __restrict__ emax,
                           float* __restrict__ yreal) {
  __shared__ float lw[KGT];
  __shared__ h1 ew[256 + KGT - 1];
  int tb = blockIdx.x % (TR / 256);
  int c = (blockIdx.x / (TR / 256)) % C_CH;
  int b = blockIdx.x / ((TR / 256) * C_CH);
  int t0 = tb * 256, tid = threadIdx.x;
  for (int i = tid; i < KGT; i += 256) lw[i] = lpf[c * KGT + i];
  for (int i = tid; i < 256 + KGT - 1; i += 256) {
    int t = t0 - 512 + i;
    ew[i] = (t >= 0 && t < TR) ? envh[((size_t)b * TR + t) * C32 + c] : (h1)0.f;
  }
  __syncthreads();
  float acc = 0.f;
  for (int k = 0; k < KGT; ++k) acc = fmaf(lw[k], (float)ew[tid + k], acc);
  float g = acc / emax[c];
  g = fminf(fmaxf(g, EPSV), 1.f);
  g = __powf(g, em1[c]);
  yreal[((size_t)b * TR + t0 + tid) * C32 + c] *= g;
}

// ------------------------ 12. peak-aligned channel sum ----------------------
__global__ void k_align_sum(const float* __restrict__ yreal, const int* __restrict__ peaks,
                            const float* __restrict__ coefp, float* __restrict__ xr) {
  __shared__ int pk[C_CH];
  int tid = threadIdx.x;
  if (tid < C_CH) pk[tid] = peaks[tid];
  __syncthreads();
  int idx = blockIdx.x * 256 + tid;                  // B*T
  int b = idx >> 17, t = idx & (T_LEN - 1);
  float s = 0.f;
#pragma unroll
  for (int c = 0; c < C_CH; ++c)
    s += yreal[((size_t)b * TR + pk[c] + t) * C32 + c];
  xr[idx] = s * coefp[0];
}

// --------------------------- 13. cochlea bwd FIR ----------------------------
__global__ void k_conv_bwd(const float* __restrict__ x, const float* __restrict__ w,
                           float* __restrict__ out) {
  __shared__ float wt[NC];
  __shared__ float xs[256 + NC - 1];
  int b = blockIdx.x >> 9;
  int t0 = (blockIdx.x & 511) << 8;
  int tid = threadIdx.x;
  const float* xb = x + (size_t)b * T_LEN;
  for (int i = tid; i < NC; i += 256) wt[i] = w[i];
  for (int i = tid; i < 256 + NC - 1; i += 256) {
    int t = t0 - PADC + i;
    xs[i] = (t >= 0 && t < T_LEN) ? xb[t] : 0.f;
  }
  __syncthreads();
  float acc = 0.f;
  for (int k = 0; k < NC; ++k) acc = fmaf(wt[k], xs[tid + k], acc);
  out[(size_t)b * T_LEN + t0 + tid] = acc;
}

// ---------------------------------------------------------------------------
extern "C" void kernel_launch(void* const* d_in, const int* in_sizes, int n_in,
                              void* d_out, int out_size, void* d_ws, size_t ws_size,
                              hipStream_t stream) {
  (void)in_sizes; (void)n_in; (void)out_size; (void)ws_size;
  const float* x     = (const float*)d_in[0];
  const float* cfwd  = (const float*)d_in[1];
  const float* cbwd  = (const float*)d_in[2];
  const float* win   = (const float*)d_in[3];
  const float* fsm   = (const float*)d_in[4];
  const float* gt    = (const float*)d_in[5];
  const float* gts   = (const float*)d_in[6];
  const float* lpf   = (const float*)d_in[7];
  const float* em1   = (const float*)d_in[8];
  const float* emax  = (const float*)d_in[9];
  const float* coef  = (const float*)d_in[10];
  const int*   peaks = (const int*)d_in[11];

  char* ws = (char*)d_ws;
  size_t off = 0;
  auto alloc = [&](size_t bytes) {
    size_t o = off; off += (bytes + 255) & ~(size_t)255; return o;
  };
  float* g_gain = (float*)(ws + alloc(B_N * sizeof(float)));
  h1* xfwd_h    = (h1*)(ws + alloc((size_t)B_N * T_LEN * 2));
  h1* bdft      = (h1*)(ws + alloc((size_t)8 * NT_STFT * 512 * 2));
  h1* bsm       = (h1*)(ws + alloc((size_t)8 * 16 * 512 * 2));
  h1* binv      = (h1*)(ws + alloc((size_t)17 * 16 * 512 * 2));
  h1* wfrag     = (h1*)(ws + alloc((size_t)4 * 33 * 512 * 2));
  h1* xsm       = (h1*)(ws + alloc((size_t)B_N * LPAD * 2));
  float* xrec   = (float*)(ws + alloc((size_t)B_N * T_LEN * 4));
  size_t region = off;

  // smear-phase buffers (alias recruitment-phase region; stream-serialized)
  size_t so = region;
  auto salloc = [&](size_t bytes) {
    size_t o = so; so += (bytes + 255) & ~(size_t)255; return o;
  };
  float* sre  = (float*)(ws + salloc((size_t)NBN * 257 * 4));
  float* sim  = (float*)(ws + salloc((size_t)NBN * 257 * 4));
  h1*    m2   = (h1*)(ws + salloc((size_t)NBN * 256 * 2));
  float* smr  = (float*)(ws + salloc((size_t)NBN * 256 * 4));
  h1*    smod = (h1*)(ws + salloc((size_t)NBN * KINV * 2));
  float* rec  = (float*)(ws + salloc((size_t)NBN * 256 * 4));

  size_t ro = region;
  auto ralloc = [&](size_t bytes) {
    size_t o = ro; ro += (bytes + 255) & ~(size_t)255; return o;
  };
  float* yreal = (float*)(ws + ralloc((size_t)B_N * TR * C32 * 4));
  h1*    envh  = (h1*)(ws + ralloc((size_t)B_N * TR * C32 * 2));

  // ----- pipeline -----
  k_calibrate<<<B_N, 256, 0, stream>>>(x, g_gain);
  k_conv_fwd<<<B_N * 512, 256, 0, stream>>>(x, cfwd, g_gain, xfwd_h);

  k_pack_gt<<<(2 * 2 * 33 * 512 + 255) / 256, 256, 0, stream>>>(gt, gts, wfrag);
  k_pack_dft<<<(8 * NT_STFT * 512 + 255) / 256, 256, 0, stream>>>(win, bdft);
  k_pack_smear<<<(8 * 16 * 512 + 255) / 256, 256, 0, stream>>>(fsm, bsm);
  k_pack_inv<<<(17 * 16 * 512 + 255) / 256, 256, 0, stream>>>(win, binv);

  k_stft<<<(MT * NT_STFT) / 8, 256, 0, stream>>>(xfwd_h, bdft, sre, sim);
  k_mag2<<<NBN, 256, 0, stream>>>(sre, sim, m2);
  k_smear_mm<<<(MT * 16) / 8, 256, 0, stream>>>(m2, bsm, smr);
  hipMemsetAsync(smod, 0, (size_t)NBN * KINV * 2, stream);
  k_ratio<<<(NBN * 257 + 255) / 256, 256, 0, stream>>>(sre, sim, smr, smod);
  k_inv<<<(MT * 16) / 8, 256, 0, stream>>>(smod, binv, rec);
  hipMemsetAsync(xsm, 0, (size_t)B_N * LPAD * 2, stream);
  k_ola<<<(B_N * T_LEN) / 256, 256, 0, stream>>>(rec, win, xsm);

  k_recruit<<<B_N * (TR / 128), 256, 0, stream>>>(xsm, wfrag, yreal, envh);
  k_lpf_gain<<<B_N * C_CH * (TR / 256), 256, 0, stream>>>(envh, lpf, em1, emax, yreal);
  k_align_sum<<<(B_N * T_LEN) / 256, 256, 0, stream>>>(yreal, peaks, coef, xrec);
  k_conv_bwd<<<B_N * 512, 256, 0, stream>>>(xrec, cbwd, (float*)d_out);
}
